// GraphModule_75007308857530
// MI455X (gfx1250) — compile-verified, matrix operand order checked by reading, not run on hardware
//
#include <hip/hip_runtime.h>
#include <math.h>

// ---------------------------------------------------------------------------
// SAGEConv (6-node graph) fused kernels for gfx1250 (MI455X).
//   y[n] = gelu( agg[n] @ W_l^T + x[n] @ W_r^T + b ),  out[b, n*256+o]
// Fused as one GEMM per node: A' = [agg | x] (B x 256 bf16),
//                             W' = [W_l ; W_r] (256 x 256 bf16, row = O).
// Memory-bound: ~600 MB total traffic @ 23.3 TB/s => ~25 us floor; bf16 WMMA
// makes the 25.8 GFLOP of matrix math negligible. Weights staged in LDS via
// CDNA5 async global->LDS copies; A fragments stream from HBM exactly once.
// ---------------------------------------------------------------------------

typedef __attribute__((ext_vector_type(16))) __bf16 v16bf;
typedef __attribute__((ext_vector_type(8)))  float  v8f;

#define NB 32768              // batch rows
#define NC 128                // input channels
#define NO 256                // output channels per node (2C)
#define NK 256                // fused K (agg|x)
#define OUT_STRIDE 1536       // 6 * 256
#define WPITCH 264            // LDS row pitch (elems): 256 + 8 pad -> +4 banks/row

__device__ __forceinline__ unsigned short f2bf(float f) {
    union { float f; unsigned u; } v; v.f = f;
    unsigned u = v.u;
    unsigned r = u + 0x7FFFu + ((u >> 16) & 1u);   // round-to-nearest-even
    return (unsigned short)(r >> 16);
}

union Frag16 { v16bf v; uint4 q[2]; };

// bf16 WMMA operand fragment: two contiguous 16B chunks (K-contiguous rows).
__device__ __forceinline__ v16bf load_frag(const unsigned short* p) {
    Frag16 f;
    f.q[0] = *(const uint4*)(p);        // K = kk .. kk+7   (this lane's half)
    f.q[1] = *(const uint4*)(p + 16);   // K = kk+16 .. kk+23
    return f.v;
}

// ---------------------------------------------------------------------------
// Pass 1a: per (b,c), mean-aggregation for all 6 nodes (hardcoded in-neighbor
// sets from the edge list) and emit bf16 A' = [agg | x].
// in-neighbors: n0:{1,2,3,4,5} n1:{0,2,3,4,5} n2:{0,1,3,4}
//               n3:{0,1,2,5}   n4:{0,1,2,5}   n5:{0,1,3,4}
// ---------------------------------------------------------------------------
__global__ __launch_bounds__(256)
void prep_x(const float* __restrict__ x, unsigned short* __restrict__ A) {
    const int i = blockIdx.x * 256 + threadIdx.x;        // over NB*NC
    const int b = i >> 7;
    const int c = i & 127;
    const int BC = NB * NC;

    const float x0 = x[0 * BC + i];
    const float x1 = x[1 * BC + i];
    const float x2 = x[2 * BC + i];
    const float x3 = x[3 * BC + i];
    const float x4 = x[4 * BC + i];
    const float x5 = x[5 * BC + i];

    const float s  = x0 + x1 + x2 + x3 + x4 + x5;
    const float a0 = (s - x0) * 0.2f;
    const float a1 = (s - x1) * 0.2f;
    const float a2 = (x0 + x1 + x3 + x4) * 0.25f;
    const float a3 = (x0 + x1 + x2 + x5) * 0.25f;
    const float a4 = (x0 + x1 + x2 + x5) * 0.25f;
    const float a5 = (x0 + x1 + x3 + x4) * 0.25f;

    size_t r;
    r = ((size_t)(0 * NB + b)) * NK + c; A[r] = f2bf(a0); A[r + NC] = f2bf(x0);
    r = ((size_t)(1 * NB + b)) * NK + c; A[r] = f2bf(a1); A[r + NC] = f2bf(x1);
    r = ((size_t)(2 * NB + b)) * NK + c; A[r] = f2bf(a2); A[r + NC] = f2bf(x2);
    r = ((size_t)(3 * NB + b)) * NK + c; A[r] = f2bf(a3); A[r + NC] = f2bf(x3);
    r = ((size_t)(4 * NB + b)) * NK + c; A[r] = f2bf(a4); A[r + NC] = f2bf(x4);
    r = ((size_t)(5 * NB + b)) * NK + c; A[r] = f2bf(a5); A[r + NC] = f2bf(x5);
}

// ---------------------------------------------------------------------------
// Pass 1b: W'[o][k] = (k < 128) ? W_l[o][k] : W_r[o][k-128], bf16.
// ---------------------------------------------------------------------------
__global__ __launch_bounds__(256)
void prep_w(const float* __restrict__ Wl, const float* __restrict__ Wr,
            unsigned short* __restrict__ W) {
    const int i = blockIdx.x * 256 + threadIdx.x;        // over NO*NK = 65536
    const int o = i >> 8;
    const int k = i & 255;
    const float v = (k < NC) ? Wl[o * NC + k] : Wr[o * NC + (k - NC)];
    W[i] = f2bf(v);
}

// ---------------------------------------------------------------------------
// Pass 2: per-node GEMM [NB x 256] x [256 x 256] via v_wmma_f32_16x16x32_bf16.
// Block: 256 thr = 8 waves; tile 128(M) x 256(N); wave = 16(M) x 256(N),
// i.e. 16 accumulators/wave. W' staged once per block into LDS with async
// global->LDS copies (ASYNCcnt); A fragments stream from global (read once).
// Fused bias + exact-erf GELU + transposed store.
// ---------------------------------------------------------------------------
__global__ __launch_bounds__(256, 2)
void sage_gemm(const unsigned short* __restrict__ A,
               const unsigned short* __restrict__ W,
               const float* __restrict__ bias,
               float* __restrict__ out) {
    __shared__ __align__(16) unsigned short sW[NO * WPITCH];   // 132 KB, 2 blk/WGP

    const int tid   = threadIdx.x;
    const int node  = blockIdx.y;
    const int mblk  = blockIdx.x;            // 128 rows each
    const int wave  = tid >> 5;              // 0..7 -> M tile
    const int lane  = tid & 31;
    const int ln16  = lane & 15;
    const int hi    = lane >> 4;             // lane half selects K sub-offset

    // --- Stage W' (256 x 256 bf16) into padded LDS via async copies. -------
    // 8192 16-byte chunks; 32 per thread.  ASYNCcnt tracks completion.
    // NOTE: LDS destination passed as the *address of sW* (ptrtoint) so the
    // compiler sees sW escape into the asm and keeps the ds_load reads live.
#pragma unroll
    for (int j = 0; j < 32; ++j) {
        const int idx   = j * 256 + tid;     // chunk id
        const int row   = idx >> 5;          // 32 chunks per 512B row
        const int chunk = idx & 31;
        const unsigned lds_byte = (unsigned)(size_t)&sW[row * WPITCH + chunk * 8];
        const unsigned short* g = W + (size_t)row * NK + chunk * 8;
        asm volatile("global_load_async_to_lds_b128 %0, %1, off"
                     :: "v"(lds_byte), "v"(g) : "memory");
    }
    asm volatile("s_wait_asynccnt 0x0" ::: "memory");
    __syncthreads();

    // --- Main K loop: 8 steps x 16 N-tiles = 128 WMMAs per wave. -----------
    const size_t rowA = (size_t)node * NB + (size_t)mblk * 128 + wave * 16 + ln16;
    const unsigned short* pA = A + rowA * NK + hi * 8;

    v8f acc[16];
    const v8f vzero = {0.f, 0.f, 0.f, 0.f, 0.f, 0.f, 0.f, 0.f};
#pragma unroll
    for (int t = 0; t < 16; ++t) acc[t] = vzero;

#pragma unroll
    for (int ks = 0; ks < 8; ++ks) {
        __builtin_prefetch(pA + (ks + 1) * 32, 0, 3);   // global_prefetch_b8
        const v16bf a = load_frag(pA + ks * 32);        // 2x global_load_b128
#pragma unroll
        for (int t = 0; t < 16; ++t) {
            // B[k][n] = W'[t*16 + n][k]; lane -> row (t*16+ln16), K-contig.
            const v16bf b = load_frag(&sW[(t * 16 + ln16) * WPITCH + ks * 32 + hi * 8]);
            acc[t] = __builtin_amdgcn_wmma_f32_16x16x32_bf16(
                /*neg_a=*/false, a, /*neg_b=*/false, b,
                /*c_mod=*/(short)0, acc[t],
                /*reuse_a=*/false, /*reuse_b=*/false);
        }
    }

    // --- Epilogue: bias + exact GELU + transposed store out[b, n*256+o]. ---
    const int mbase = mblk * 128 + wave * 16 + hi * 8;   // lanes16-31: M = r+8
#pragma unroll
    for (int t = 0; t < 16; ++t) {
        const int ocol = t * 16 + ln16;
        const float bb = bias[ocol];
        const int col = node * NO + ocol;
#pragma unroll
        for (int r = 0; r < 8; ++r) {
            float v = acc[t][r] + bb;
            v = 0.5f * v * (1.0f + erff(v * 0.70710678118654752f));
            out[(size_t)(mbase + r) * OUT_STRIDE + col] = v;
        }
    }
}

// ---------------------------------------------------------------------------
extern "C" void kernel_launch(void* const* d_in, const int* in_sizes, int n_in,
                              void* d_out, int out_size, void* d_ws, size_t ws_size,
                              hipStream_t stream) {
    (void)in_sizes; (void)n_in; (void)out_size; (void)ws_size;
    const float* x    = (const float*)d_in[0];   // [6, 32768, 128]
    const float* Wl   = (const float*)d_in[1];   // [256, 128]
    const float* Wr   = (const float*)d_in[2];   // [256, 128]
    const float* bias = (const float*)d_in[3];   // [256]
    float* out = (float*)d_out;                  // [32768, 1536]

    unsigned short* Abf = (unsigned short*)d_ws;                 // 6*NB*NK bf16 (~100.7 MB)
    unsigned short* Wbf = Abf + (size_t)6 * NB * NK;             // NO*NK bf16 (128 KB)

    // Pass 1: bf16 conversion + aggregation + weight concat.
    prep_x<<<dim3((NB * NC) / 256), 256, 0, stream>>>(x, Abf);
    prep_w<<<dim3((NO * NK) / 256), 256, 0, stream>>>(Wl, Wr, Wbf);

    // Pass 2: fused WMMA GEMM + bias + GELU + transpose-store.
    sage_gemm<<<dim3(NB / 128, 6), 256, 0, stream>>>(Abf, Wbf, bias, out);
}
